// MultiHeadsSelfAttention_23192823399166
// MI455X (gfx1250) — matrix operand order repaired and checked by reading, not compiled, not run on hardware
//
#include <hip/hip_runtime.h>
#include <hip/hip_bf16.h>

// Problem constants (from reference)
#define BATCH 2
#define SEQ   2048
#define CH    1024
#define NH    16
#define HD    64
#define MTOT  (BATCH * SEQ)   // 4096 rows for the big GEMMs

typedef __bf16 bf16_t;
typedef bf16_t v16bf __attribute__((ext_vector_type(16)));
typedef float  v8f   __attribute__((ext_vector_type(8)));

union FragAB { v16bf v; unsigned short u[16]; uint4 q[2]; };
union FragC  { v8f v; float f[8]; };

__device__ __forceinline__ unsigned short f32_to_bf16_raw(float f) {
  unsigned int u = __float_as_uint(f);
  u += 0x7FFFu + ((u >> 16) & 1u);   // round-to-nearest-even
  return (unsigned short)(u >> 16);
}

// ---------------------------------------------------------------------------
// fp32 -> bf16 conversion (vectorized 4/thread)
// ---------------------------------------------------------------------------
__global__ void cvt_f32_to_bf16(const float* __restrict__ in,
                                unsigned short* __restrict__ out, int n) {
  int i = (blockIdx.x * blockDim.x + threadIdx.x) * 4;
  if (i + 3 < n) {
    float4 f = *reinterpret_cast<const float4*>(in + i);
    uint2 p;
    p.x = (unsigned)f32_to_bf16_raw(f.x) | ((unsigned)f32_to_bf16_raw(f.y) << 16);
    p.y = (unsigned)f32_to_bf16_raw(f.z) | ((unsigned)f32_to_bf16_raw(f.w) << 16);
    *reinterpret_cast<uint2*>(out + i) = p;
  }
}

// ---------------------------------------------------------------------------
// GEMM: y[m][o] = sum_c A[m][c] * W[o][c] + bias[o]
//   A: [4096][1024] bf16 row-major, W: [1024][1024] bf16 row-major.
// 32x64 output tile per wave (2 A-frags x 4 B-frags -> 8 WMMAs / 12 loads).
//   store_mode 0: bf16 output scattered to [B,H,N,D]  (QKV projections)
//   store_mode 1: fp32 output row-major [4096][1024]  (final projection)
// Grid: 2048 wave tiles (128 Mtiles x 16 Ntiles) / 8 waves = 256 blocks.
// ---------------------------------------------------------------------------
__global__ void __launch_bounds__(256)
gemm_bf16_wmma(const unsigned short* __restrict__ A,
               const unsigned short* __restrict__ W,
               const float* __restrict__ bias,
               unsigned short* __restrict__ out_bhnd,
               float* __restrict__ out_f32,
               int store_mode) {
  const int lane = threadIdx.x & 31;
  const int wave = threadIdx.x >> 5;
  const int tile = blockIdx.x * 8 + wave;
  const int mt   = tile >> 4;        // 0..127 (rows of 32)
  const int nt   = tile & 15;        // 0..15  (cols of 64)
  const int half = lane >> 4;
  const int lidx = lane & 15;

  const unsigned short* arow[2];
#pragma unroll
  for (int ms = 0; ms < 2; ++ms)
    arow[ms] = A + (size_t)(mt * 32 + ms * 16 + lidx) * CH;
  const unsigned short* brow[4];
#pragma unroll
  for (int ns = 0; ns < 4; ++ns)
    brow[ns] = W + (size_t)(nt * 64 + ns * 16 + lidx) * CH;

  FragC acc[2][4];
#pragma unroll
  for (int ms = 0; ms < 2; ++ms)
#pragma unroll
    for (int ns = 0; ns < 4; ++ns)
#pragma unroll
      for (int i = 0; i < 8; ++i) acc[ms][ns].f[i] = 0.0f;

#pragma unroll 2
  for (int kb = 0; kb < CH; kb += 32) {
    FragAB a[2], b[4];
#pragma unroll
    for (int ms = 0; ms < 2; ++ms) {
      a[ms].q[0] = *reinterpret_cast<const uint4*>(arow[ms] + kb + 16 * half);
      a[ms].q[1] = *reinterpret_cast<const uint4*>(arow[ms] + kb + 16 * half + 8);
    }
#pragma unroll
    for (int ns = 0; ns < 4; ++ns) {
      b[ns].q[0] = *reinterpret_cast<const uint4*>(brow[ns] + kb + 16 * half);
      b[ns].q[1] = *reinterpret_cast<const uint4*>(brow[ns] + kb + 16 * half + 8);
    }
#pragma unroll
    for (int ms = 0; ms < 2; ++ms)
#pragma unroll
      for (int ns = 0; ns < 4; ++ns)
        acc[ms][ns].v = __builtin_amdgcn_wmma_f32_16x16x32_bf16(
            false, a[ms].v, false, b[ns].v, (short)0, acc[ms][ns].v, false, false);
  }

  float bval[4];
#pragma unroll
  for (int ns = 0; ns < 4; ++ns) bval[ns] = bias[nt * 64 + ns * 16 + lidx];

  if (store_mode == 0) {
#pragma unroll
    for (int ms = 0; ms < 2; ++ms)
#pragma unroll
      for (int ns = 0; ns < 4; ++ns)
#pragma unroll
        for (int r = 0; r < 8; ++r) {
          int row = mt * 32 + ms * 16 + r + 8 * half;   // n_tot
          int col = nt * 64 + ns * 16 + lidx;           // h*HD + d
          float val = acc[ms][ns].f[r] + bval[ns];
          int bb = row >> 11, nseq = row & (SEQ - 1);
          int h = col >> 6, d = col & (HD - 1);
          out_bhnd[(((size_t)(bb * NH + h)) * SEQ + nseq) * HD + d] =
              f32_to_bf16_raw(val);
        }
  } else {
#pragma unroll
    for (int ms = 0; ms < 2; ++ms)
#pragma unroll
      for (int ns = 0; ns < 4; ++ns)
#pragma unroll
        for (int r = 0; r < 8; ++r) {
          int row = mt * 32 + ms * 16 + r + 8 * half;
          int col = nt * 64 + ns * 16 + lidx;
          out_f32[(size_t)row * CH + col] = acc[ms][ns].f[r] + bval[ns];
        }
  }
}

// ---------------------------------------------------------------------------
// Flash-style attention. Block = 8 waves = 128 query rows of one (b,h).
// Per 32-key block: K tile (32x64) staged to LDS with CDNA5 async-copy,
// V tile staged transposed (64x32) so B-fragments are contiguous ds_load_b128.
// Online softmax in registers; P relaid out C->A layout through per-wave LDS.
// ---------------------------------------------------------------------------
__global__ void __launch_bounds__(256)
attn_fwd_wmma(const unsigned short* __restrict__ Q,
              const unsigned short* __restrict__ K,
              const unsigned short* __restrict__ V,
              unsigned short* __restrict__ ctx) {
  __shared__ __attribute__((aligned(16))) unsigned short Ktile[32 * 64];     // 4 KB
  __shared__ __attribute__((aligned(16))) unsigned short Vt[64 * 32];        // 4 KB (transposed)
  __shared__ __attribute__((aligned(16))) unsigned short Ptile[8][16 * 32];  // 8 KB

  const int tid  = threadIdx.x;
  const int lane = tid & 31;
  const int wave = tid >> 5;
  const int half = lane >> 4;
  const int lidx = lane & 15;

  const int rb = blockIdx.x & 15;     // 16 row-blocks of 128 query rows
  const int bh = blockIdx.x >> 4;     // 0..31
  const int b  = bh >> 4;
  const int h  = bh & 15;

  const int q0 = rb * 128 + wave * 16;
  const unsigned short* qbase = Q + ((size_t)bh * SEQ + q0) * HD;
  const unsigned short* kbase = K + (size_t)bh * SEQ * HD;
  const unsigned short* vbase = V + (size_t)bh * SEQ * HD;

  // Q A-fragments: 16x64 = two K-chunks of 32
  FragAB qa[2];
#pragma unroll
  for (int c = 0; c < 2; ++c) {
    const unsigned short* r = qbase + (size_t)lidx * HD + c * 32;
    qa[c].q[0] = *reinterpret_cast<const uint4*>(r + 16 * half);
    qa[c].q[1] = *reinterpret_cast<const uint4*>(r + 16 * half + 8);
  }

  FragC accO[4];
#pragma unroll
  for (int t = 0; t < 4; ++t)
#pragma unroll
    for (int i = 0; i < 8; ++i) accO[t].f[i] = 0.0f;

  float mrow[8], lrow[8];
#pragma unroll
  for (int i = 0; i < 8; ++i) { mrow[i] = -3.0e38f; lrow[i] = 0.0f; }

  const float scale = 0.125f;   // 1/sqrt(64)

  for (int k0 = 0; k0 < SEQ; k0 += 32) {
    // ---- cooperative staging of K (async copy) and V (manual transpose) ----
    {
      // K: 32x64 tile is 4 KB contiguous; thread tid moves 16 bytes.
      unsigned lds_k = (unsigned)(uintptr_t)(&Ktile[tid * 8]);
      unsigned long long gk =
          (unsigned long long)(uintptr_t)(kbase + (size_t)k0 * HD + tid * 8);
      asm volatile("global_load_async_to_lds_b128 %0, %1, off"
                   :: "v"(lds_k), "v"(gk) : "memory");

      // V: load 8 contiguous elems of row (k0+vrow), scatter into Vt[d][k].
      int vrow = tid >> 3, vseg = tid & 7;
      uint4 vv = *reinterpret_cast<const uint4*>(
          vbase + (size_t)(k0 + vrow) * HD + vseg * 8);
      const unsigned short* vs = reinterpret_cast<const unsigned short*>(&vv);
#pragma unroll
      for (int e = 0; e < 8; ++e)
        Vt[(vseg * 8 + e) * 32 + vrow] = vs[e];

      asm volatile("s_wait_asynccnt 0" ::: "memory");
    }
    __syncthreads();

    // ---- S = Q * K^T for a 16x32 score tile (two 16-key column tiles) ----
    FragC s[2];
#pragma unroll
    for (int ct = 0; ct < 2; ++ct) {
#pragma unroll
      for (int i = 0; i < 8; ++i) s[ct].f[i] = 0.0f;
#pragma unroll
      for (int kc = 0; kc < 2; ++kc) {
        FragAB bk;   // B[d][n] = K[key n][d]; 32 contiguous bytes in LDS
        const unsigned short* r =
            &Ktile[(ct * 16 + lidx) * 64 + kc * 32 + 16 * half];
        bk.q[0] = *reinterpret_cast<const uint4*>(r);
        bk.q[1] = *reinterpret_cast<const uint4*>(r + 8);
        s[ct].v = __builtin_amdgcn_wmma_f32_16x16x32_bf16(
            false, qa[kc].v, false, bk.v, (short)0, s[ct].v, false, false);
      }
    }

    // ---- online softmax over the 32 new columns, per row r ----
#pragma unroll
    for (int r = 0; r < 8; ++r) {
      float s0 = s[0].f[r] * scale;
      float s1 = s[1].f[r] * scale;
      float mx = fmaxf(s0, s1);
#pragma unroll
      for (int off = 1; off < 16; off <<= 1)
        mx = fmaxf(mx, __shfl_xor(mx, off, 32));      // stays within half-wave
      float mnew  = fmaxf(mrow[r], mx);
      float alpha = __expf(mrow[r] - mnew);
      float p0 = __expf(s0 - mnew);
      float p1 = __expf(s1 - mnew);
      float rs = p0 + p1;
#pragma unroll
      for (int off = 1; off < 16; off <<= 1)
        rs += __shfl_xor(rs, off, 32);
      lrow[r] = lrow[r] * alpha + rs;
      mrow[r] = mnew;
#pragma unroll
      for (int t = 0; t < 4; ++t) accO[t].f[r] *= alpha;

      int m = r + 8 * half;                           // C-layout row
      Ptile[wave][m * 32 + lidx]      = f32_to_bf16_raw(p0);
      Ptile[wave][m * 32 + 16 + lidx] = f32_to_bf16_raw(p1);
    }

    // LDS ops from one wave are in-order; wait + clobber pins the relayout
    asm volatile("s_wait_dscnt 0" ::: "memory");

    // P as A-fragment (16x32 bf16) out of LDS
    FragAB pa;
    pa.q[0] = *reinterpret_cast<const uint4*>(&Ptile[wave][lidx * 32 + 16 * half]);
    pa.q[1] = *reinterpret_cast<const uint4*>(&Ptile[wave][lidx * 32 + 16 * half + 8]);

    // ---- O += P * V : V B-frags are contiguous 32B reads from Vt ----
#pragma unroll
    for (int t = 0; t < 4; ++t) {
      FragAB bv;   // element j <-> K = 16*half + j at column d = t*16 + lidx
      const unsigned short* r = &Vt[(t * 16 + lidx) * 32 + 16 * half];
      bv.q[0] = *reinterpret_cast<const uint4*>(r);
      bv.q[1] = *reinterpret_cast<const uint4*>(r + 8);
      accO[t].v = __builtin_amdgcn_wmma_f32_16x16x32_bf16(
          false, pa.v, false, bv.v, (short)0, accO[t].v, false, false);
    }
    __syncthreads();   // protect K/V tiles before next iteration overwrites
  }

  // ---- normalize and store context in [n_tot][C] layout ----
#pragma unroll
  for (int r = 0; r < 8; ++r) {
    float inv = 1.0f / lrow[r];
    int row = q0 + r + 8 * half;
    int ntot = b * SEQ + row;
#pragma unroll
    for (int t = 0; t < 4; ++t) {
      int col = h * HD + t * 16 + lidx;
      ctx[(size_t)ntot * CH + col] = f32_to_bf16_raw(accO[t].f[r] * inv);
    }
  }
}

// ---------------------------------------------------------------------------
// Host-side orchestration
// ---------------------------------------------------------------------------
extern "C" void kernel_launch(void* const* d_in, const int* in_sizes, int n_in,
                              void* d_out, int out_size, void* d_ws, size_t ws_size,
                              hipStream_t stream) {
  (void)in_sizes; (void)n_in; (void)out_size; (void)ws_size;

  const float* x  = (const float*)d_in[0];
  const float* Wq = (const float*)d_in[1];
  const float* bq = (const float*)d_in[2];
  const float* Wk = (const float*)d_in[3];
  const float* bk = (const float*)d_in[4];
  const float* Wv = (const float*)d_in[5];
  const float* bv = (const float*)d_in[6];
  const float* Wo = (const float*)d_in[7];
  const float* bo = (const float*)d_in[8];
  float* out = (float*)d_out;

  unsigned char* ws = (unsigned char*)d_ws;
  const size_t MB = 1u << 20;
  unsigned short* xb   = (unsigned short*)(ws + 0 * MB);    // 8 MB  [4096][1024]
  unsigned short* wqb  = (unsigned short*)(ws + 8 * MB);    // 2 MB
  unsigned short* wkb  = (unsigned short*)(ws + 10 * MB);   // 2 MB
  unsigned short* wvb  = (unsigned short*)(ws + 12 * MB);   // 2 MB
  unsigned short* wob  = (unsigned short*)(ws + 14 * MB);   // 2 MB
  unsigned short* qb   = (unsigned short*)(ws + 16 * MB);   // 8 MB  [B*H][SEQ][HD]
  unsigned short* kbuf = (unsigned short*)(ws + 24 * MB);   // 8 MB
  unsigned short* vbuf = (unsigned short*)(ws + 32 * MB);   // 8 MB
  unsigned short* ctxb = (unsigned short*)(ws + 40 * MB);   // 8 MB  [4096][1024]

  // 1) convert activations + weights to bf16
  cvt_f32_to_bf16<<<4096, 256, 0, stream>>>(x,  xb,  MTOT * CH);
  cvt_f32_to_bf16<<<1024, 256, 0, stream>>>(Wq, wqb, CH * CH);
  cvt_f32_to_bf16<<<1024, 256, 0, stream>>>(Wk, wkb, CH * CH);
  cvt_f32_to_bf16<<<1024, 256, 0, stream>>>(Wv, wvb, CH * CH);
  cvt_f32_to_bf16<<<1024, 256, 0, stream>>>(Wo, wob, CH * CH);

  // 2) QKV projections (x @ W^T + b), stored head-split [B,H,N,D] bf16
  gemm_bf16_wmma<<<256, 256, 0, stream>>>(xb, wqb, bq, qb,   nullptr, 0);
  gemm_bf16_wmma<<<256, 256, 0, stream>>>(xb, wkb, bk, kbuf, nullptr, 0);
  gemm_bf16_wmma<<<256, 256, 0, stream>>>(xb, wvb, bv, vbuf, nullptr, 0);

  // 3) flash attention, context written back as [B*N][C] bf16
  attn_fwd_wmma<<<512, 256, 0, stream>>>(qb, kbuf, vbuf, ctxb);

  // 4) output projection (ctx @ Wo^T + bo) -> fp32 d_out
  gemm_bf16_wmma<<<256, 256, 0, stream>>>(ctxb, wob, bo, nullptr, out, 1);
}
